// IterativeVoxelModel_84507776516323
// MI455X (gfx1250) — compile-verified, matrix operand order checked by reading, not run on hardware
//
#include <hip/hip_runtime.h>

typedef __attribute__((ext_vector_type(16))) _Float16 v16h;
typedef __attribute__((ext_vector_type(8)))  float    v8f;
typedef __attribute__((ext_vector_type(4)))  float    f32x4;

#define B_TOT   32768
#define DIM     64
#define NHEAD   4
#define HDIM    16
#define NLAYER  3
#define NN      27
#define CTR     13
#define EPSF    1e-5f
#define TILE_B  16
#define ROWS    (TILE_B * NN)   /* 432 */
#define THREADS 256             /* 8 wave32 -> 2 waves per SIMD32 */
#define ATT_SCALE 0.25f         /* HD^-0.5 */

// ---------------- LDS sizing (element counts) ----------------
constexpr int SZ_X   = ROWS * DIM;      // 27648 halves
constexpr int SZ_KV  = ROWS * DIM;      // 27648 halves each
constexpr int SZ_W64 = DIM * DIM;       // 4096 halves (Wq/Wk/Wv/Wo)
constexpr int SZ_W1  = DIM * 4 * DIM;   // 16384 halves
constexpr int SZ_W2  = 4 * DIM * DIM;   // 16384 halves
constexpr int SZ_T   = TILE_B * DIM;    // 1024 halves
constexpr int SZ_H   = TILE_B * 4*DIM;  // 4096 halves

constexpr size_t SMEM_HALF  = (size_t)SZ_X + 2*SZ_KV + 4*SZ_W64 + SZ_W1 + SZ_W2
                            + SZ_T /*tgt*/ + SZ_T /*o*/ + SZ_H;
constexpr size_t SMEM_F32   = 1024 /*q*/ + 1024 /*center*/ + 8*64 + 256 + 64;
constexpr size_t SMEM_BYTES = SMEM_HALF*2 + SMEM_F32*4 + 16*4;

// ---------------- WMMA fragment helpers ----------------
__device__ inline v16h make_frag(f32x4 a, f32x4 b) {
  union { f32x4 q[2]; v16h v; } u;
  u.q[0] = a; u.q[1] = b;
  return u.v;
}

// A fragment (16x32 f16) from row-major LDS: lane L<16 -> row L, K in {0..7,16..23};
// lane L>=16 -> row L-16, K in {8..15,24..31}. Two 128-bit LDS loads.
__device__ inline v16h frag_a(const _Float16* base, int row0, int k0, int ld, int lane) {
  int r = lane & 15, hi = lane >> 4;
  const _Float16* p = base + (size_t)(row0 + r) * ld + k0 + hi * 8;
  return make_frag(*(const f32x4*)p, *(const f32x4*)(p + 16));
}

// B fragment from pre-swizzled weight storage: contiguous 16 halves per lane.
__device__ inline v16h frag_b(const _Float16* sw, int kt, int nt, int NT, int lane) {
  const _Float16* p = sw + ((size_t)((kt * NT + nt) << 5) + lane) * 16;
  return make_frag(*(const f32x4*)p, *(const f32x4*)(p + 8));
}

__device__ inline v8f wmma_f16(v16h a, v16h b, v8f c) {
  return __builtin_amdgcn_wmma_f32_16x16x32_f16(false, a, false, b, (short)0, c,
                                                false, false);
}

// Stage a row-major f32 weight (Kdim x Ndim) into LDS as f16 in B-fragment layout.
__device__ inline void stage_sw(const float* __restrict__ g, _Float16* dst,
                                int Kdim, int Ndim, int tid) {
  const int NT = Ndim >> 4;
  const int total = Kdim * Ndim;
  for (int e = tid; e < total; e += THREADS) {
    int K  = e / Ndim, nf = e - K * Ndim;
    int nt = nf >> 4,  n  = nf & 15;
    int kt = K >> 5,   kk = K & 31;
    int q2 = kk >> 3;
    int hi = q2 & 1;
    int h  = (kk & 7) | ((q2 & 2) << 2);
    dst[((size_t)((kt * NT + nt) << 5) + n + (hi << 4)) * 16 + h] = (_Float16)g[e];
  }
}

__device__ inline void copyf(const float* __restrict__ g, float* dst, int n, int tid) {
  for (int i = tid; i < n; i += THREADS) dst[i] = g[i];
}

// ---------------- main fused kernel ----------------
__global__ __launch_bounds__(THREADS, 1)
void iterative_voxel_kernel(
    const float* __restrict__ patch, const float* __restrict__ kmask,
    const float* __restrict__ w_in,  const float* __restrict__ b_in,
    const float* __restrict__ pos,
    const float* __restrict__ Wq, const float* __restrict__ bq,
    const float* __restrict__ Wk, const float* __restrict__ bk,
    const float* __restrict__ Wv, const float* __restrict__ bv,
    const float* __restrict__ Wo, const float* __restrict__ bo,
    const float* __restrict__ g1, const float* __restrict__ be1,
    const float* __restrict__ g2, const float* __restrict__ be2,
    const float* __restrict__ W1, const float* __restrict__ b1,
    const float* __restrict__ W2, const float* __restrict__ b2,
    const float* __restrict__ Wh, const float* __restrict__ bh,
    float* __restrict__ out)
{
  extern __shared__ __align__(16) char smem_raw[];
  _Float16* sX  = (_Float16*)smem_raw;          // (432,64) x as f16
  _Float16* sK  = sX  + SZ_X;                   // (432,64) k
  _Float16* sV  = sK  + SZ_KV;                  // (432,64) v
  _Float16* sWq = sV  + SZ_KV;                  // swizzled weights
  _Float16* sWk = sWq + SZ_W64;
  _Float16* sWv = sWk + SZ_W64;
  _Float16* sWo = sWv + SZ_W64;
  _Float16* sW1 = sWo + SZ_W64;
  _Float16* sW2 = sW1 + SZ_W1;
  _Float16* sT  = sW2 + SZ_W2;                  // (16,64) tgt / t2
  _Float16* sO  = sT  + SZ_T;                   // (16,64) attention output
  _Float16* sH  = sO  + SZ_T;                   // (16,256) MLP hidden
  float* sQ  = (float*)(sH + SZ_H);             // (16,64) q
  float* sC  = sQ  + 1024;                      // (16,64) center, f32
  float* sBq = sC  + 1024;
  float* sBk = sBq + 64;
  float* sBv = sBk + 64;
  float* sBo = sBv + 64;
  float* sG1 = sBo + 64;
  float* sE1 = sG1 + 64;
  float* sG2 = sE1 + 64;
  float* sE2 = sG2 + 64;
  float* sB1 = sE2 + 64;                        // 256
  float* sB2 = sB1 + 256;                       // 64
  int*   sMk = (int*)(sB2 + 64);                // 16 mask bitfields

  const int tid  = threadIdx.x;
  const int wv   = tid >> 5;
  const int lane = tid & 31;
  const int b0   = blockIdx.x * TILE_B;

  // ---- build x (f16), center (f32), mask bits ----
  for (int e = tid; e < TILE_B * NN * DIM; e += THREADS) {
    int b = e / (NN * DIM); int rem = e - b * (NN * DIM);
    int n = rem >> 6;       int d   = rem & 63;
    float pv  = patch[(size_t)(b0 + b) * NN + n];
    float val = pv * w_in[d] + b_in[d] + pos[n * DIM + d];
    sX[e] = (_Float16)val;
    if (n == CTR) sC[b * DIM + d] = val;
  }
  if (tid < TILE_B) {
    int bits = 0;
    for (int n = 0; n < NN; ++n)
      if (kmask[(size_t)(b0 + tid) * NN + n] > 0.5f) bits |= (1 << n);
    sMk[tid] = bits;
  }
  __syncthreads();

  for (int l = 0; l < NLAYER; ++l) {
    // ---- stage this layer's weights / vectors ----
    stage_sw(Wq + l*SZ_W64, sWq, DIM, DIM, tid);
    stage_sw(Wk + l*SZ_W64, sWk, DIM, DIM, tid);
    stage_sw(Wv + l*SZ_W64, sWv, DIM, DIM, tid);
    stage_sw(Wo + l*SZ_W64, sWo, DIM, DIM, tid);
    stage_sw(W1 + l*SZ_W1,  sW1, DIM, 4*DIM, tid);
    stage_sw(W2 + l*SZ_W2,  sW2, 4*DIM, DIM, tid);
    copyf(bq  + l*DIM,   sBq, DIM, tid);
    copyf(bk  + l*DIM,   sBk, DIM, tid);
    copyf(bv  + l*DIM,   sBv, DIM, tid);
    copyf(bo  + l*DIM,   sBo, DIM, tid);
    copyf(g1  + l*DIM,   sG1, DIM, tid);
    copyf(be1 + l*DIM,   sE1, DIM, tid);
    copyf(g2  + l*DIM,   sG2, DIM, tid);
    copyf(be2 + l*DIM,   sE2, DIM, tid);
    copyf(b1  + l*4*DIM, sB1, 4*DIM, tid);
    copyf(b2  + l*DIM,   sB2, DIM, tid);
    if (l + 1 < NLAYER) {  // L2 prefetch of next layer's big weights
      const float* n1 = W1 + (l+1)*SZ_W1;
      const float* n2 = W2 + (l+1)*SZ_W2;
      for (int i = tid * 32; i < SZ_W1; i += THREADS * 32) {
        __builtin_prefetch(n1 + i, 0, 1);
        __builtin_prefetch(n2 + i, 0, 1);
      }
    }
    __syncthreads();

    // ---- LN1: tgt = LN(center)*g1+be1 (one batch row per thread) ----
    if (tid < TILE_B) {
      const float* c = sC + tid * DIM;
      float m = 0.f;
      for (int d = 0; d < DIM; ++d) m += c[d];
      m *= (1.f / DIM);
      float va = 0.f;
      for (int d = 0; d < DIM; ++d) { float t = c[d] - m; va += t * t; }
      va *= (1.f / DIM);
      float r = rsqrtf(va + EPSF);
      for (int d = 0; d < DIM; ++d)
        sT[tid * DIM + d] = (_Float16)((c[d] - m) * r * sG1[d] + sE1[d]);
    }
    __syncthreads();

    // ---- q = tgt @ Wq + bq : waves 0-3 (one 16-col tile each) ----
    if (wv < 4) {
      v16h a0 = frag_a(sT, 0, 0,  DIM, lane);
      v16h a1 = frag_a(sT, 0, 32, DIM, lane);
      v8f acc = {0,0,0,0,0,0,0,0};
      acc = wmma_f16(a0, frag_b(sWq, 0, wv, 4, lane), acc);
      acc = wmma_f16(a1, frag_b(sWq, 1, wv, 4, lane), acc);
      int n = lane & 15, hi = lane >> 4, col = (wv << 4) + n;
      for (int j = 0; j < 8; ++j)
        sQ[(j + (hi << 3)) * DIM + col] = acc[j] + sBq[col];
    }
    // ---- k,v = X(432x64) @ Wk/Wv over 8 waves ----
    // wave w handles column tile nt = w&3, mt strided by 2 starting at 1-(w>>2);
    // all 4 B fragments are loop-invariant and stay pinned in VGPRs.
    {
      const int nt  = wv & 3;
      const int mt0 = wv >> 2;          // waves 0-3: odd extra Q work, start mt 0? no:
      v16h bk0 = frag_b(sWk, 0, nt, 4, lane);
      v16h bk1 = frag_b(sWk, 1, nt, 4, lane);
      v16h bv0 = frag_b(sWv, 0, nt, 4, lane);
      v16h bv1 = frag_b(sWv, 1, nt, 4, lane);
      const int n = lane & 15, hi = lane >> 4, col = (nt << 4) + n;
      const float biask = sBk[col], biasv = sBv[col];
      // waves 0-3 (which also did Q) take mt = 1,3,5,... (13 tiles);
      // waves 4-7 take mt = 0,2,4,... (14 tiles)
      for (int mt = (mt0 ? 0 : 1); mt < 27; mt += 2) {
        v16h a0 = frag_a(sX, mt << 4, 0,  DIM, lane);
        v16h a1 = frag_a(sX, mt << 4, 32, DIM, lane);
        v8f ak = {0,0,0,0,0,0,0,0};
        v8f av = {0,0,0,0,0,0,0,0};
        ak = wmma_f16(a0, bk0, ak);
        ak = wmma_f16(a1, bk1, ak);
        av = wmma_f16(a0, bv0, av);
        av = wmma_f16(a1, bv1, av);
        for (int j = 0; j < 8; ++j) {
          int row = (mt << 4) + j + (hi << 3);
          sK[row * DIM + col] = (_Float16)(ak[j] + biask);
          sV[row * DIM + col] = (_Float16)(av[j] + biasv);
        }
      }
    }
    __syncthreads();

    // ---- attention: one (batch, head) per thread ----
    if (tid < TILE_B * NHEAD) {
      int b = tid >> 2, h = tid & 3;
      float q[HDIM];
      for (int d = 0; d < HDIM; ++d) q[d] = sQ[b * DIM + h * HDIM + d];
      int bits = sMk[b];
      float sc[NN];
      float mx = -1e30f;
      for (int n = 0; n < NN; ++n) {
        float s;
        if (bits == 0) s = 0.f;                 // has_any == false -> uniform
        else if ((bits >> n) & 1) {
          s = 0.f;
          const _Float16* kp = sK + (size_t)(b * NN + n) * DIM + h * HDIM;
          for (int d = 0; d < HDIM; ++d) s += q[d] * (float)kp[d];
          s *= ATT_SCALE;
        } else s = -1e30f;                      // masked -> -inf
        sc[n] = s; mx = fmaxf(mx, s);
      }
      float sum = 0.f;
      for (int n = 0; n < NN; ++n) { float e = __expf(sc[n] - mx); sc[n] = e; sum += e; }
      float inv = 1.f / sum;
      float o[HDIM];
      for (int d = 0; d < HDIM; ++d) o[d] = 0.f;
      for (int n = 0; n < NN; ++n) {
        float w = sc[n] * inv;
        const _Float16* vp = sV + (size_t)(b * NN + n) * DIM + h * HDIM;
        for (int d = 0; d < HDIM; ++d) o[d] += w * (float)vp[d];
      }
      for (int d = 0; d < HDIM; ++d)
        sO[b * DIM + h * HDIM + d] = (_Float16)o[d];
    }
    __syncthreads();

    // ---- center += o @ Wo + bo : waves 0-3 ----
    if (wv < 4) {
      v16h a0 = frag_a(sO, 0, 0,  DIM, lane);
      v16h a1 = frag_a(sO, 0, 32, DIM, lane);
      v8f acc = {0,0,0,0,0,0,0,0};
      acc = wmma_f16(a0, frag_b(sWo, 0, wv, 4, lane), acc);
      acc = wmma_f16(a1, frag_b(sWo, 1, wv, 4, lane), acc);
      int n = lane & 15, hi = lane >> 4, col = (wv << 4) + n;
      for (int j = 0; j < 8; ++j)
        sC[(j + (hi << 3)) * DIM + col] += acc[j] + sBo[col];
    }
    __syncthreads();

    // ---- LN2 -> sT ----
    if (tid < TILE_B) {
      const float* c = sC + tid * DIM;
      float m = 0.f;
      for (int d = 0; d < DIM; ++d) m += c[d];
      m *= (1.f / DIM);
      float va = 0.f;
      for (int d = 0; d < DIM; ++d) { float t = c[d] - m; va += t * t; }
      va *= (1.f / DIM);
      float r = rsqrtf(va + EPSF);
      for (int d = 0; d < DIM; ++d)
        sT[tid * DIM + d] = (_Float16)((c[d] - m) * r * sG2[d] + sE2[d]);
    }
    __syncthreads();

    // ---- MLP1: h = gelu(t2 @ W1 + b1), 16 col tiles over 8 waves ----
    {
      v16h a0 = frag_a(sT, 0, 0,  DIM, lane);
      v16h a1 = frag_a(sT, 0, 32, DIM, lane);
      for (int nt = wv; nt < 16; nt += 8) {
        v8f acc = {0,0,0,0,0,0,0,0};
        acc = wmma_f16(a0, frag_b(sW1, 0, nt, 16, lane), acc);
        acc = wmma_f16(a1, frag_b(sW1, 1, nt, 16, lane), acc);
        int n = lane & 15, hi = lane >> 4, col = (nt << 4) + n;
        for (int j = 0; j < 8; ++j) {
          float x = acc[j] + sB1[col];
          float g = 0.5f * x * (1.f + erff(x * 0.70710678118654752f));
          sH[(size_t)(j + (hi << 3)) * (4 * DIM) + col] = (_Float16)g;
        }
      }
    }
    __syncthreads();

    // ---- MLP2: center += h @ W2 + b2 : waves 0-3, 8 K steps each ----
    if (wv < 4) {
      v8f acc = {0,0,0,0,0,0,0,0};
      for (int kt = 0; kt < 8; ++kt) {
        v16h a = frag_a(sH, 0, kt << 5, 4 * DIM, lane);
        acc = wmma_f16(a, frag_b(sW2, kt, wv, 4, lane), acc);
      }
      int n = lane & 15, hi = lane >> 4, col = (wv << 4) + n;
      for (int j = 0; j < 8; ++j)
        sC[(j + (hi << 3)) * DIM + col] += acc[j] + sB2[col];
    }
    __syncthreads();

    // ---- write center back into x[:,CTR,:] ----
    for (int e = tid; e < TILE_B * DIM; e += THREADS) {
      int b = e >> 6, d = e & 63;
      sX[(size_t)(b * NN + CTR) * DIM + d] = (_Float16)sC[e];
    }
    __syncthreads();
  }

  // ---- head: out = center @ Wh + bh ----
  if (tid < TILE_B) {
    float acc = bh[0];
    for (int d = 0; d < DIM; ++d) acc += sC[tid * DIM + d] * Wh[d];
    out[b0 + tid] = acc;
  }
}

extern "C" void kernel_launch(void* const* d_in, const int* in_sizes, int n_in,
                              void* d_out, int out_size, void* d_ws, size_t ws_size,
                              hipStream_t stream) {
  (void)in_sizes; (void)n_in; (void)out_size; (void)d_ws; (void)ws_size;
  const float* patch = (const float*)d_in[0];
  const float* kmask = (const float*)d_in[1];
  const float* w_in  = (const float*)d_in[2];
  const float* b_in  = (const float*)d_in[3];
  const float* pos   = (const float*)d_in[4];
  const float* Wq    = (const float*)d_in[5];
  const float* bq    = (const float*)d_in[6];
  const float* Wk    = (const float*)d_in[7];
  const float* bk    = (const float*)d_in[8];
  const float* Wv    = (const float*)d_in[9];
  const float* bv    = (const float*)d_in[10];
  const float* Wo    = (const float*)d_in[11];
  const float* bo    = (const float*)d_in[12];
  const float* g1    = (const float*)d_in[13];
  const float* be1   = (const float*)d_in[14];
  const float* g2    = (const float*)d_in[15];
  const float* be2   = (const float*)d_in[16];
  const float* W1    = (const float*)d_in[17];
  const float* b1    = (const float*)d_in[18];
  const float* W2    = (const float*)d_in[19];
  const float* b2    = (const float*)d_in[20];
  const float* Wh    = (const float*)d_in[21];
  const float* bh    = (const float*)d_in[22];
  float* out = (float*)d_out;

  (void)hipFuncSetAttribute((const void*)iterative_voxel_kernel,
                            hipFuncAttributeMaxDynamicSharedMemorySize,
                            (int)SMEM_BYTES);

  dim3 grid(B_TOT / TILE_B);   // 2048 workgroups, 16 batch elements each
  dim3 block(THREADS);         // 8 wave32
  iterative_voxel_kernel<<<grid, block, SMEM_BYTES, stream>>>(
      patch, kmask, w_in, b_in, pos, Wq, bq, Wk, bk, Wv, bv, Wo, bo,
      g1, be1, g2, be2, W1, b1, W2, b2, Wh, bh, out);
}
